// MultiHeadAttention_68487548502817
// MI455X (gfx1250) — matrix operand order repaired and checked by reading, not compiled, not run on hardware
//
#include <hip/hip_runtime.h>

#define L_SEQ   2048
#define D_MODEL 1024
#define H_HEADS 16
#define HD      64
#define WIN     256

typedef __attribute__((ext_vector_type(16))) _Float16 v16h;
typedef __attribute__((ext_vector_type(8)))  _Float16 h8;
typedef __attribute__((ext_vector_type(4)))  _Float16 h4;
typedef __attribute__((ext_vector_type(8)))  float    v8f;
typedef __attribute__((ext_vector_type(4)))  int      v4i;

#if defined(__gfx1250__) && __has_builtin(__builtin_amdgcn_global_load_async_to_lds_b128)
#define HAVE_ASYNC_LDS 1
#else
#define HAVE_ASYNC_LDS 0
#endif

__device__ __forceinline__ v16h cat16(h8 lo, h8 hi) {
  return __builtin_shufflevector(lo, hi, 0,1,2,3,4,5,6,7,8,9,10,11,12,13,14,15);
}
__device__ __forceinline__ v8f wmma16(v16h a, v16h b, v8f c) {
  // v_wmma_f32_16x16x32_f16: D = A(16x32 f16) * B(32x16 f16) + C(16x16 f32)
  return __builtin_amdgcn_wmma_f32_16x16x32_f16(false, a, false, b, (short)0, c, false, false);
}
__device__ __forceinline__ int imin(int a, int b) { return a < b ? a : b; }
__device__ __forceinline__ int imax(int a, int b) { return a > b ? a : b; }

// 16-byte global -> LDS copy; async (ASYNCcnt) when available.
__device__ __forceinline__ void async_b128(const _Float16* g, _Float16* l) {
#if HAVE_ASYNC_LDS
  typedef __attribute__((address_space(1))) v4i gvec;
  typedef __attribute__((address_space(3))) v4i lvec;
  __builtin_amdgcn_global_load_async_to_lds_b128((gvec*)g, (lvec*)l,
                                                 /*offset=*/0, /*cpol=*/0);
#else
  *(uint4*)l = *(const uint4*)g;
#endif
}
__device__ __forceinline__ void wait_async() {
#if HAVE_ASYNC_LDS
#if __has_builtin(__builtin_amdgcn_s_wait_asynccnt)
  __builtin_amdgcn_s_wait_asynccnt(0);
#else
  asm volatile("s_wait_asynccnt 0" ::: "memory");
#endif
#endif
}

// ---------------------------------------------------------------------------
// f32 -> f16 conversion (vectorized by 4)
// ---------------------------------------------------------------------------
__global__ void cvt_f32_to_f16_kernel(const float* __restrict__ in,
                                      _Float16* __restrict__ out, int n) {
  int i = (blockIdx.x * blockDim.x + threadIdx.x) * 4;
  if (i + 3 < n) {
    float4 f = *(const float4*)(in + i);
    h4 h = { (_Float16)f.x, (_Float16)f.y, (_Float16)f.z, (_Float16)f.w };
    *(h4*)(out + i) = h;
  }
}

// ---------------------------------------------------------------------------
// Tiled WMMA GEMM: C[M,N] = A[M,K] * Bw[N,K]^T + bias[N]
// 64x64 macro tile, 4 waves/block, each wave computes a 32x32 sub-tile
// (2x2 WMMA tiles). K stepped by 32 with double-buffered async LDS staging:
// issue tile k+1 (global_load_async_to_lds_b128) -> WMMA tile k ->
// s_wait_asynccnt 0 -> barrier.
// ---------------------------------------------------------------------------
template <bool HALF_OUT>
__global__ __launch_bounds__(128)
void gemm_wmma_kernel(const _Float16* __restrict__ A,   // [M,K]
                      const _Float16* __restrict__ Bw,  // [N,K] (i.e. B^T)
                      const float* __restrict__ bias,   // [N]
                      void* __restrict__ Cout,
                      int M, int N, int K) {
  __shared__ _Float16 As[2][64][40];   // padded stride: 80B rows
  __shared__ _Float16 Bs[2][64][40];
  const int tid  = threadIdx.x;
  const int wave = tid >> 5, lane = tid & 31;
  const int m0 = blockIdx.y * 64, n0 = blockIdx.x * 64;
  const int wm = (wave & 1) * 32, wn = (wave >> 1) * 32;
  const int fm   = lane & 15;
  const int kb8  = (lane >> 4) * 8;    // A-frag K base (interleaved halves)
  const int kb16 = (lane >> 4) * 16;   // contiguous K base (unused here; B-frag uses kb8)
  const int lrow = tid >> 1;           // cooperative-load row 0..63
  const int lk   = (tid & 1) * 16;     // cooperative-load k offset
  v8f acc[2][2] = {};

  auto issue = [&](int buf, int k0) {
    const _Float16* ga = A  + (size_t)(m0 + lrow) * K + k0 + lk;
    const _Float16* gb = Bw + (size_t)(n0 + lrow) * K + k0 + lk;
    async_b128(ga,     &As[buf][lrow][lk]);
    async_b128(ga + 8, &As[buf][lrow][lk + 8]);
    async_b128(gb,     &Bs[buf][lrow][lk]);
    async_b128(gb + 8, &Bs[buf][lrow][lk + 8]);
  };

  issue(0, 0);
  wait_async();
  __syncthreads();

  int buf = 0;
  for (int k0 = 0; k0 < K; k0 += 32, buf ^= 1) {
    if (k0 + 32 < K) issue(buf ^ 1, k0 + 32);   // prefetch next tile (overlaps WMMA)
    v16h af[2], bf[2];
#pragma unroll
    for (int t = 0; t < 2; ++t) {
      const _Float16* pa = &As[buf][wm + t * 16 + fm][0];
      af[t] = cat16(*(const h8*)(pa + kb8), *(const h8*)(pa + kb8 + 16));
      const _Float16* pb = &Bs[buf][wn + t * 16 + fm][0];
      bf[t] = cat16(*(const h8*)(pb + kb8), *(const h8*)(pb + kb8 + 16));
    }
    acc[0][0] = wmma16(af[0], bf[0], acc[0][0]);
    acc[0][1] = wmma16(af[0], bf[1], acc[0][1]);
    acc[1][0] = wmma16(af[1], bf[0], acc[1][0]);
    acc[1][1] = wmma16(af[1], bf[1], acc[1][1]);
    wait_async();      // next tile resident
    __syncthreads();   // all waves done reading current tile
  }

  const int cm = (lane >> 4) * 8, cn = lane & 15;
#pragma unroll
  for (int i = 0; i < 2; ++i)
#pragma unroll
    for (int j = 0; j < 2; ++j) {
      int gn = n0 + wn + j * 16 + cn;
      float bv = bias[gn];
#pragma unroll
      for (int v = 0; v < 8; ++v) {
        int gm = m0 + wm + i * 16 + cm + v;
        float val = acc[i][j][v] + bv;
        if (HALF_OUT) ((_Float16*)Cout)[(size_t)gm * N + gn] = (_Float16)val;
        else          ((float*)Cout)[(size_t)gm * N + gn] = val;
      }
    }
}

// ---------------------------------------------------------------------------
// Sliding-window attention, flash-style online softmax.
// One wave handles one (b, h, 16-query tile); keys processed in 32-chunks.
// V chunk staged via async global->LDS copies that overlap the S-tile WMMAs.
// qkv: [B*L, 3*D] f16 (bias already applied). outh: [B*L, D] f16.
// ---------------------------------------------------------------------------
__global__ __launch_bounds__(128)
void attn_wmma_kernel(const _Float16* __restrict__ qkv,
                      _Float16* __restrict__ outh) {
  __shared__ _Float16 Pbuf[4][16][40];   // per-wave P (16 q x 32 keys, padded)
  __shared__ _Float16 Vbuf[4][32][72];   // per-wave V (32 keys x 64 feats, padded)
  const int tid  = threadIdx.x;
  const int wave = tid >> 5, lane = tid & 31;
  const int gw = blockIdx.x * 4 + wave;            // 0..4095
  const int qt = gw & (L_SEQ / 16 - 1);            // query tile 0..127
  const int bh = gw >> 7;                          // 0..31
  const int b = bh >> 4, h = bh & 15;
  const int q0 = qt * 16;
  const size_t rs3 = 3 * D_MODEL;
  const _Float16* Qp = qkv + (size_t)b * L_SEQ * rs3 + h * HD;
  const _Float16* Kp = Qp + D_MODEL;
  const _Float16* Vp = Qp + 2 * D_MODEL;
  const int fm   = lane & 15;
  const int kb8  = (lane >> 4) * 8;
  const int kb16 = (lane >> 4) * 16;
  const int cm   = (lane >> 4) * 8, cn = lane & 15;

  // Q A-fragments for feature halves 0..31 and 32..63
  v16h qa[2];
#pragma unroll
  for (int t = 0; t < 2; ++t) {
    const _Float16* p = Qp + (size_t)(q0 + fm) * rs3 + t * 32 + kb8;
    qa[t] = cat16(*(const h8*)p, *(const h8*)(p + 16));
  }

  float m_i[8], l_i[8];
#pragma unroll
  for (int v = 0; v < 8; ++v) { m_i[v] = -1e30f; l_i[v] = 0.0f; }
  v8f o[4] = {};

  const int kstart = imax(0, q0 - WIN) & ~31;
  const int kend   = imin(L_SEQ, q0 + 15 + WIN + 1);

  for (int kc = kstart; kc < kend; kc += 32) {
    // stage V chunk asynchronously: lane -> key row kc+lane (clamped)
    {
      int krow = imin(kc + lane, L_SEQ - 1);
      const _Float16* vp = Vp + (size_t)krow * rs3;
      _Float16* dst = &Vbuf[wave][lane][0];
#pragma unroll
      for (int j = 0; j < 8; ++j) async_b128(vp + j * 8, dst + j * 8);
    }
    // S = Q * K^T for two 16-key subtiles (overlaps the V copy)
    v8f s[2];
#pragma unroll
    for (int u = 0; u < 2; ++u) {
      int kn = imin(kc + u * 16 + fm, L_SEQ - 1);
      const _Float16* kp = Kp + (size_t)kn * rs3;
      v8f c = {};
#pragma unroll
      for (int t = 0; t < 2; ++t) {
        const _Float16* p = kp + t * 32 + kb8;
        c = wmma16(qa[t], cat16(*(const h8*)p, *(const h8*)(p + 16)), c);
      }
      s[u] = c;
    }
    // masked online softmax; row stats reduced across 16-lane halves
    float alpha[8];
#pragma unroll
    for (int v = 0; v < 8; ++v) {
      const int gq = q0 + cm + v;
      const int gk0 = kc + cn, gk1 = kc + 16 + cn;
      float s0 = (gk0 < L_SEQ && gk0 >= gq - WIN && gk0 <= gq + WIN)
                     ? s[0][v] * 0.125f : -1e30f;
      float s1 = (gk1 < L_SEQ && gk1 >= gq - WIN && gk1 <= gq + WIN)
                     ? s[1][v] * 0.125f : -1e30f;
      float mc = fmaxf(s0, s1);
#pragma unroll
      for (int off = 8; off; off >>= 1) mc = fmaxf(mc, __shfl_xor(mc, off, 16));
      float mn = fmaxf(m_i[v], mc);
      alpha[v] = __expf(m_i[v] - mn);
      float e0 = __expf(s0 - mn), e1 = __expf(s1 - mn);
      float rsum = e0 + e1;
#pragma unroll
      for (int off = 8; off; off >>= 1) rsum += __shfl_xor(rsum, off, 16);
      l_i[v] = l_i[v] * alpha[v] + rsum;
      m_i[v] = mn;
      Pbuf[wave][cm + v][cn]      = (_Float16)e0;
      Pbuf[wave][cm + v][16 + cn] = (_Float16)e1;
    }
    asm volatile("" ::: "memory");   // keep LDS writes before reads (per-wave DS is in-order)
    wait_async();                    // V chunk resident in LDS
    // rescale running output, then O += P * V
#pragma unroll
    for (int nt = 0; nt < 4; ++nt)
#pragma unroll
      for (int v = 0; v < 8; ++v) o[nt][v] = o[nt][v] * alpha[v];
    v16h pa;
    {
      const _Float16* pp = &Pbuf[wave][fm][0];
      pa = cat16(*(const h8*)(pp + kb16), *(const h8*)(pp + kb16 + 8));
    }
#pragma unroll
    for (int nt = 0; nt < 4; ++nt) {
      v16h vb;
#pragma unroll
      for (int i2 = 0; i2 < 16; ++i2) vb[i2] = Vbuf[wave][kb16 + i2][nt * 16 + fm];
      o[nt] = wmma16(pa, vb, o[nt]);
    }
    asm volatile("" ::: "memory");
  }

  _Float16* op = outh + (size_t)(b * L_SEQ + q0) * D_MODEL + h * HD;
#pragma unroll
  for (int nt = 0; nt < 4; ++nt)
#pragma unroll
    for (int v = 0; v < 8; ++v) {
      float val = o[nt][v] / l_i[v];
      op[(size_t)(cm + v) * D_MODEL + nt * 16 + cn] = (_Float16)val;
    }
}

// ---------------------------------------------------------------------------
// Launcher
// ---------------------------------------------------------------------------
extern "C" void kernel_launch(void* const* d_in, const int* in_sizes, int n_in,
                              void* d_out, int out_size, void* d_ws, size_t ws_size,
                              hipStream_t stream) {
  (void)in_sizes; (void)n_in; (void)out_size; (void)ws_size;
  const float* x     = (const float*)d_in[0];
  const float* qkv_w = (const float*)d_in[1];
  const float* qkv_b = (const float*)d_in[2];
  const float* out_w = (const float*)d_in[3];
  const float* out_b = (const float*)d_in[4];
  float* out = (float*)d_out;

  const int B = 2, L = L_SEQ, D = D_MODEL;
  const int M = B * L;  // 4096 tokens

  char* ws = (char*)d_ws;
  size_t off = 0;
  _Float16* Xh    = (_Float16*)(ws + off); off += (size_t)M * D * 2;        // 8 MB
  _Float16* Wqkvh = (_Float16*)(ws + off); off += (size_t)3 * D * D * 2;    // 6 MB
  _Float16* Woh   = (_Float16*)(ws + off); off += (size_t)D * D * 2;        // 2 MB
  _Float16* QKVh  = (_Float16*)(ws + off); off += (size_t)M * 3 * D * 2;    // 24 MB
  _Float16* AOh   = (_Float16*)(ws + off);                                  // 8 MB

  const int nX = M * D, nWq = 3 * D * D, nWo = D * D;
  cvt_f32_to_f16_kernel<<<(nX / 4 + 255) / 256, 256, 0, stream>>>(x, Xh, nX);
  cvt_f32_to_f16_kernel<<<(nWq / 4 + 255) / 256, 256, 0, stream>>>(qkv_w, Wqkvh, nWq);
  cvt_f32_to_f16_kernel<<<(nWo / 4 + 255) / 256, 256, 0, stream>>>(out_w, Woh, nWo);

  // QKV projection: [4096,1024] x [3072,1024]^T -> f16 [4096,3072]
  gemm_wmma_kernel<true><<<dim3(3 * D / 64, M / 64), 128, 0, stream>>>(
      Xh, Wqkvh, qkv_b, (void*)QKVh, M, 3 * D, D);

  // Sliding-window attention: 4096 waves (B*H*L/16), 4 waves per block
  attn_wmma_kernel<<<(B * H_HEADS * (L / 16)) / 4, 128, 0, stream>>>(QKVh, AOh);

  // Output projection: [4096,1024] x [1024,1024]^T + bias -> f32 d_out
  gemm_wmma_kernel<false><<<dim3(D / 64, M / 64), 128, 0, stream>>>(
      AOh, Woh, out_b, (void*)out, M, D, D);
}
